// Attention_29618094473506
// MI455X (gfx1250) — matrix operand order repaired and checked by reading, not compiled, not run on hardware
//
#include <hip/hip_runtime.h>
#include <hip/hip_bf16.h>

// ---------------------------------------------------------------------------
// Multi-head attention forward for B=4, S=2048, D=1024, H=16, HD=64.
// All matmuls in bf16 via V_WMMA_F32_16X16X32_BF16 (f32 accumulate).
//   - projections: 64x64 output tile per wave (16 WMMA / K-step, A,B reused 4x)
//   - attention : flash-style; workgroup = 8 waves x 16 q-rows of one head,
//                 K/V 64-key chunks double-buffered through LDS (8x traffic cut),
//                 staged with GLOBAL_LOAD_ASYNC_TO_LDS_B128 when available
//   - score scale sqrt(HD)=8 folded into the Q projection epilogue
// ---------------------------------------------------------------------------

typedef __bf16 bf16_t;
typedef __attribute__((ext_vector_type(16))) __bf16 v16bf;
typedef __attribute__((ext_vector_type(8)))  __bf16 v8bf;
typedef __attribute__((ext_vector_type(8)))  float  v8f;
typedef __attribute__((ext_vector_type(4)))  int    v4i;

#define S_LEN 2048
#define D_DIM 1024
#define N_HEAD 16
#define HD 64
#define B_SZ 4
#define M_ROWS (B_SZ * S_LEN)   // 8192
#define PSTRIDE 72              // LDS row stride (bf16): 144B, 16B aligned

// ---- optional CDNA5 async global->LDS path (ASYNCcnt) ---------------------
#if defined(__has_builtin)
#if __has_builtin(__builtin_amdgcn_global_load_async_to_lds_b128)
#define HAVE_ASYNC_LDS 1
#endif
#endif

#ifdef HAVE_ASYNC_LDS
typedef __attribute__((address_space(1))) v4i* gas_v4i;   // global v4i*
typedef __attribute__((address_space(3))) v4i* las_v4i;   // LDS v4i*
#endif

__device__ __forceinline__ void wait_async_all() {
#ifdef HAVE_ASYNC_LDS
#if __has_builtin(__builtin_amdgcn_s_wait_asynccnt)
    __builtin_amdgcn_s_wait_asynccnt(0);
#else
    asm volatile("s_wait_asynccnt 0" ::: "memory");
#endif
#endif
}

union FragBF { v16bf v; v8bf h[2]; };

__device__ __forceinline__ __bf16 to_bf16(float x) { return (__bf16)x; }

__device__ __forceinline__ v8f vzero8() {
    v8f z;
#pragma unroll
    for (int i = 0; i < 8; ++i) z[i] = 0.0f;
    return z;
}

// D = A*B + C : 16x16 f32 out, A 16x32 bf16, B 32x16 bf16
__device__ __forceinline__ v8f wmma_bf16(v16bf a, v16bf b, v8f c) {
    return __builtin_amdgcn_wmma_f32_16x16x32_bf16(
        /*neg_a=*/false, a, /*neg_b=*/false, b,
        /*c_mod=*/(short)0, c, /*reuse_a=*/false, /*reuse_b=*/false);
}

// A-fragment (16x32 bf16, row-major source, row stride ld elements).
// Lane L: row m=L&15 ; elems 0..7 = K(8*hi+0..7), elems 8..15 = K(16+8*hi+0..7)
__device__ __forceinline__ v16bf frag_a_bf16(const bf16_t* base, int ld) {
    const int lane = threadIdx.x & 31;
    const int m = lane & 15, hi = lane >> 4;
    const bf16_t* p = base + (size_t)m * ld + 8 * hi;
    FragBF f;
    f.h[0] = *(const v8bf*)(p);
    f.h[1] = *(const v8bf*)(p + 16);
    return f.v;
}

// Same fragment but source is f32; convert to bf16 on the fly.
__device__ __forceinline__ v16bf frag_a_f32(const float* base, int ld) {
    const int lane = threadIdx.x & 31;
    const int m = lane & 15, hi = lane >> 4;
    const float* p = base + (size_t)m * ld + 8 * hi;
    FragBF f;
#pragma unroll
    for (int i = 0; i < 8; ++i) f.v[i] = to_bf16(p[i]);
#pragma unroll
    for (int i = 0; i < 8; ++i) f.v[8 + i] = to_bf16(p[16 + i]);
    return f.v;
}

// B-fragment (32x16 bf16). Source stored "N-major": row n holds the 32 K
// values of column n contiguously. Lane L: col n=L&15 ; elems 0..7 = K(8*hi+0..7), elems 8..15 = K(16+8*hi+0..7).
__device__ __forceinline__ v16bf frag_b_bf16(const bf16_t* base, int ld) {
    const int lane = threadIdx.x & 31;
    const int n = lane & 15, hi = lane >> 4;
    const bf16_t* p = base + (size_t)n * ld + 8 * hi;
    FragBF f;
    f.h[0] = *(const v8bf*)(p);
    f.h[1] = *(const v8bf*)(p + 16);
    return f.v;
}

// ---------------------------------------------------------------------------
// f32 -> bf16 conversion (weights)
// ---------------------------------------------------------------------------
__global__ void cvt_f32_to_bf16(const float* __restrict__ src,
                                bf16_t* __restrict__ dst, int n) {
    int i = blockIdx.x * blockDim.x + threadIdx.x;
    if (i < n) dst[i] = to_bf16(src[i]);
}

// ---------------------------------------------------------------------------
// GEMM: C[M,N] = (A[M,K] * W[N,K]^T + bias[N]) * scale
//   A_F32: 1 -> A is f32 (convert on the fly), 0 -> A is bf16
//   MODE : 0 -> store bf16 [M,N]
//          1 -> store bf16 transposed as Vt[b*H+h][hd][S]  (N split as h*64+hd)
//          2 -> store f32 [M,N]
// One wave computes a 64x64 tile: 4 A-frags x 4 B-frags -> 16 WMMA per K-step.
// ---------------------------------------------------------------------------
template <int A_F32, int MODE>
__global__ __launch_bounds__(256) void gemm_bf16_wmma(
    const void* __restrict__ Ap, const bf16_t* __restrict__ W,
    const float* __restrict__ bias, void* __restrict__ Op,
    int M, int N, int K, float scale) {
    const int lane = threadIdx.x & 31;
    const int wid  = blockIdx.x * (blockDim.x >> 5) + (threadIdx.x >> 5);
    const int nStrips = N >> 6;                 // tiles of 64 cols
    const int mt = wid / nStrips;
    const int nt = wid - mt * nStrips;
    if (mt * 64 >= M) return;
    const int mbase = mt * 64, nbase = nt * 64;
    const int nl = lane & 15, hi = lane >> 4;

    v8f acc[4][4];                              // [m-tile][n-tile]
#pragma unroll
    for (int mi = 0; mi < 4; ++mi)
#pragma unroll
        for (int ni = 0; ni < 4; ++ni) acc[mi][ni] = vzero8();

    for (int k0 = 0; k0 < K; k0 += 32) {
        // prefetch next K-tiles (lowers to global_prefetch_b8)
        if (k0 + 64 < K) {
            __builtin_prefetch(W + (size_t)(nbase + nl) * K + k0 + 64, 0, 1);
            if (A_F32)
                __builtin_prefetch((const float*)Ap + (size_t)(mbase + nl) * K + k0 + 64, 0, 1);
            else
                __builtin_prefetch((const bf16_t*)Ap + (size_t)(mbase + nl) * K + k0 + 64, 0, 1);
        }
        v16bf a[4];
#pragma unroll
        for (int mi = 0; mi < 4; ++mi) {
            if (A_F32)
                a[mi] = frag_a_f32((const float*)Ap + (size_t)(mbase + 16 * mi) * K + k0, K);
            else
                a[mi] = frag_a_bf16((const bf16_t*)Ap + (size_t)(mbase + 16 * mi) * K + k0, K);
        }
#pragma unroll
        for (int ni = 0; ni < 4; ++ni) {
            v16bf b = frag_b_bf16(W + (size_t)(nbase + 16 * ni) * K + k0, K);
#pragma unroll
            for (int mi = 0; mi < 4; ++mi)
                acc[mi][ni] = wmma_bf16(a[mi], b, acc[mi][ni]);
        }
    }

#pragma unroll
    for (int ni = 0; ni < 4; ++ni) {
        const int col = nbase + 16 * ni + nl;
        const float bia = bias[col];
#pragma unroll
        for (int mi = 0; mi < 4; ++mi) {
#pragma unroll
            for (int r = 0; r < 8; ++r) {
                const int row = mbase + 16 * mi + r + 8 * hi;
                const float val = (acc[mi][ni][r] + bia) * scale;
                if (MODE == 0) {
                    ((bf16_t*)Op)[(size_t)row * N + col] = to_bf16(val);
                } else if (MODE == 1) {
                    const int bb = row >> 11, ss = row & (S_LEN - 1);
                    const int hh = col >> 6,  dd = col & (HD - 1);
                    ((bf16_t*)Op)[(((size_t)(bb * N_HEAD + hh) * HD + dd) << 11) + ss] =
                        to_bf16(val);
                } else {
                    ((float*)Op)[(size_t)row * N + col] = val;
                }
            }
        }
    }
}

// ---------------------------------------------------------------------------
// Flash attention, LDS-cooperative:
//   block = 256 threads = 8 waves; owns one (b,h) and 128 consecutive q rows
//   (16 rows per wave). K/V 64-key chunks are double-buffered in LDS and
//   shared by all 8 waves (one __syncthreads per chunk). When the toolchain
//   exposes them, chunks are staged with GLOBAL_LOAD_ASYNC_TO_LDS_B128
//   (ASYNCcnt) instead of a VGPR round-trip.
//   Q   : bf16 [B*S, D]   (pre-scaled by sqrt(HD) in the projection)
//   Kt  : bf16 [B*S, D]   -> LDS tile [key][dim]   (frag_b rows = keys)
//   Vt  : bf16 [B*H][HD][S] -> LDS tile [dim][key] (frag_b rows = dims)
//   Out : bf16 [B*S, D]
// ---------------------------------------------------------------------------
__global__ __launch_bounds__(256) void flash_attn_wmma(
    const bf16_t* __restrict__ Q, const bf16_t* __restrict__ Kt,
    const bf16_t* __restrict__ Vt, bf16_t* __restrict__ Out) {
    __shared__ __align__(16) bf16_t Kl[2][64 * PSTRIDE];   // [key][dim]
    __shared__ __align__(16) bf16_t Vl[2][64 * PSTRIDE];   // [dim][key]
    __shared__ __align__(16) bf16_t Pl[8][16 * PSTRIDE];   // per-wave P tile

    const int tid  = threadIdx.x;
    const int lane = tid & 31;
    const int wv   = tid >> 5;
    const int bh   = blockIdx.x >> 4;          // b*H + h
    const int qg   = blockIdx.x & 15;          // 128-row q group within head
    const int b    = bh >> 4, h = bh & (N_HEAD - 1);
    const int q0   = qg * 128 + wv * 16;

    const bf16_t* qp = Q  + ((size_t)(b * S_LEN + q0)) * D_DIM + h * HD;
    const bf16_t* kp = Kt + ((size_t)(b * S_LEN)) * D_DIM + h * HD;
    const bf16_t* vp = Vt + ((size_t)bh * HD) * S_LEN;

    // cooperative loader: 64x64 bf16 K tile + 64x64 bf16 V tile, 16B per op
    auto load_chunk = [&](int j, int buf) {
#pragma unroll
        for (int c = tid; c < 512; c += 256) {
            const int row = c >> 3;            // 0..63
            const int col = (c & 7) << 3;      // 0,8,..,56
            const bf16_t* gk = kp + (size_t)(j + row) * D_DIM + col;
            const bf16_t* gv = vp + (size_t)row * S_LEN + j + col;
            bf16_t* lk = &Kl[buf][row * PSTRIDE + col];
            bf16_t* lv = &Vl[buf][row * PSTRIDE + col];
#ifdef HAVE_ASYNC_LDS
            // AS(1)/AS(3) v4i pointers built via integer casts; LDS generic
            // addresses carry the offset in addr[31:0] (ISA aperture rule).
            __builtin_amdgcn_global_load_async_to_lds_b128(
                (gas_v4i)(unsigned long long)gk,
                (las_v4i)(unsigned int)(unsigned long long)lk, 0, 0);
            __builtin_amdgcn_global_load_async_to_lds_b128(
                (gas_v4i)(unsigned long long)gv,
                (las_v4i)(unsigned int)(unsigned long long)lv, 0, 0);
#else
            *(v8bf*)lk = *(const v8bf*)gk;
            *(v8bf*)lv = *(const v8bf*)gv;
#endif
        }
    };

    // Q tile (16x64) lives in registers: two 16x32 A-fragments.
    const v16bf aq0 = frag_a_bf16(qp, D_DIM);
    const v16bf aq1 = frag_a_bf16(qp + 32, D_DIM);

    v8f accO[4];
#pragma unroll
    for (int t = 0; t < 4; ++t) accO[t] = vzero8();
    float mrow[8], lrow[8];
#pragma unroll
    for (int r = 0; r < 8; ++r) { mrow[r] = -3.0e38f; lrow[r] = 0.0f; }

    bf16_t* lp = &Pl[wv][0];
    const int hi = lane >> 4, nl = lane & 15;

    load_chunk(0, 0);
    wait_async_all();
    __syncthreads();

    for (int jn = 0; jn < S_LEN / 64; ++jn) {
        const int cur = jn & 1;
        if (jn + 1 < S_LEN / 64) load_chunk((jn + 1) * 64, cur ^ 1);

        const bf16_t* kl = &Kl[cur][0];
        const bf16_t* vl = &Vl[cur][0];

        // ---- scores: S = Q (16x64) x K^T (64x64), four 16x16 tiles ------
        v8f sc[4];
#pragma unroll
        for (int t = 0; t < 4; ++t) {
            sc[t] = vzero8();
            sc[t] = wmma_bf16(aq0, frag_b_bf16(kl + 16 * t * PSTRIDE, PSTRIDE), sc[t]);
            sc[t] = wmma_bf16(aq1, frag_b_bf16(kl + 16 * t * PSTRIDE + 32, PSTRIDE), sc[t]);
        }

        // ---- online softmax (rows: M = r + 8*hi, cols across 16 lanes) --
#pragma unroll
        for (int r = 0; r < 8; ++r) {
            const float x0 = sc[0][r], x1 = sc[1][r];
            const float x2 = sc[2][r], x3 = sc[3][r];
            float mx = fmaxf(fmaxf(x0, x1), fmaxf(x2, x3));
            for (int d = 1; d < 16; d <<= 1)
                mx = fmaxf(mx, __shfl_xor(mx, d, 32));
            const float mnew = fmaxf(mrow[r], mx);
            const float corr = __expf(mrow[r] - mnew);
            const float p0 = __expf(x0 - mnew), p1 = __expf(x1 - mnew);
            const float p2 = __expf(x2 - mnew), p3 = __expf(x3 - mnew);
            float sum = (p0 + p1) + (p2 + p3);
            for (int d = 1; d < 16; d <<= 1)
                sum += __shfl_xor(sum, d, 32);
            lrow[r] = lrow[r] * corr + sum;
            mrow[r] = mnew;
#pragma unroll
            for (int t = 0; t < 4; ++t) accO[t][r] *= corr;
            const int ro = (r + 8 * hi) * PSTRIDE;
            lp[ro + nl]      = to_bf16(p0);
            lp[ro + 16 + nl] = to_bf16(p1);
            lp[ro + 32 + nl] = to_bf16(p2);
            lp[ro + 48 + nl] = to_bf16(p3);
        }

        // ---- O += P (16x64) x V (64x64) ---------------------------------
        const v16bf ap0 = frag_a_bf16(lp, PSTRIDE);
        const v16bf ap1 = frag_a_bf16(lp + 32, PSTRIDE);
#pragma unroll
        for (int t = 0; t < 4; ++t) {
            accO[t] = wmma_bf16(ap0, frag_b_bf16(vl + 16 * t * PSTRIDE, PSTRIDE), accO[t]);
            accO[t] = wmma_bf16(ap1, frag_b_bf16(vl + 16 * t * PSTRIDE + 32, PSTRIDE), accO[t]);
        }

        wait_async_all();  // next chunk's DMA done (in-order => this chunk too)
        __syncthreads();   // chunk jn fully consumed; buffer may be refilled
    }

    // ---- normalize and store [B*S, D] bf16 -----------------------------
    bf16_t* op = Out + ((size_t)(b * S_LEN + q0)) * D_DIM + h * HD;
#pragma unroll
    for (int t = 0; t < 4; ++t) {
#pragma unroll
        for (int r = 0; r < 8; ++r) {
            op[(size_t)(r + 8 * hi) * D_DIM + 16 * t + nl] =
                to_bf16(accO[t][r] / lrow[r]);
        }
    }
}

// ---------------------------------------------------------------------------
// Host-side orchestration
// ---------------------------------------------------------------------------
extern "C" void kernel_launch(void* const* d_in, const int* in_sizes, int n_in,
                              void* d_out, int out_size, void* d_ws, size_t ws_size,
                              hipStream_t stream) {
    const float* q_in = (const float*)d_in[0];
    const float* k_in = (const float*)d_in[1];
    const float* v_in = (const float*)d_in[2];
    const float* Wq = (const float*)d_in[3];
    const float* bq = (const float*)d_in[4];
    const float* Wk = (const float*)d_in[5];
    const float* bk = (const float*)d_in[6];
    const float* Wv = (const float*)d_in[7];
    const float* bv = (const float*)d_in[8];
    const float* Wp = (const float*)d_in[9];
    const float* bp = (const float*)d_in[10];

    // workspace layout (bf16 elements): 4x 2MB weights + 4x 16MB buffers = 72MB
    bf16_t* ws   = (bf16_t*)d_ws;
    const size_t WN = (size_t)D_DIM * D_DIM;
    const size_t AN = (size_t)M_ROWS * D_DIM;
    bf16_t* wqb = ws;
    bf16_t* wkb = wqb + WN;
    bf16_t* wvb = wkb + WN;
    bf16_t* wpb = wvb + WN;
    bf16_t* qbuf = wpb + WN;
    bf16_t* kbuf = qbuf + AN;
    bf16_t* vtbf = kbuf + AN;
    bf16_t* abuf = vtbf + AN;

    // 1) weights f32 -> bf16
    {
        const int n = (int)WN, blk = 256, g = (n + blk - 1) / blk;
        cvt_f32_to_bf16<<<g, blk, 0, stream>>>(Wq, wqb, n);
        cvt_f32_to_bf16<<<g, blk, 0, stream>>>(Wk, wkb, n);
        cvt_f32_to_bf16<<<g, blk, 0, stream>>>(Wv, wvb, n);
        cvt_f32_to_bf16<<<g, blk, 0, stream>>>(Wp, wpb, n);
    }

    // 2) projections: (8192/64)*(1024/64) = 2048 waves / 8 = 256 blocks
    const int gemm_blocks = (M_ROWS / 64) * (D_DIM / 64) / 8;
    gemm_bf16_wmma<1, 0><<<gemm_blocks, 256, 0, stream>>>(
        q_in, wqb, bq, qbuf, M_ROWS, D_DIM, D_DIM, 8.0f);   // fold sqrt(HD)
    gemm_bf16_wmma<1, 0><<<gemm_blocks, 256, 0, stream>>>(
        k_in, wkb, bk, kbuf, M_ROWS, D_DIM, D_DIM, 1.0f);
    gemm_bf16_wmma<1, 1><<<gemm_blocks, 256, 0, stream>>>(
        v_in, wvb, bv, vtbf, M_ROWS, D_DIM, D_DIM, 1.0f);

    // 3) attention: 64 (b,h) x 16 q-groups = 1024 blocks x 256 threads
    flash_attn_wmma<<<B_SZ * N_HEAD * (S_LEN / 128), 256, 0, stream>>>(
        qbuf, kbuf, vtbf, abuf);

    // 4) output projection -> f32 d_out
    gemm_bf16_wmma<0, 2><<<gemm_blocks, 256, 0, stream>>>(
        abuf, wpb, bp, d_out, M_ROWS, D_DIM, D_DIM, 1.0f);
}
